// HSTv5_23124103922219
// MI455X (gfx1250) — compile-verified, matrix operand order checked by reading, not run on hardware
//
#include <hip/hip_runtime.h>
#include <hip/hip_bf16.h>
#include <stdint.h>

// ---------------- problem constants (from reference) ----------------
#define Bsz 4
#define Ssz 2048
#define Dd  1024
#define Ff  4096
#define Hh  16
#define HDh 64
#define BS  (Bsz * Ssz)   // 8192 rows
#define CPB 32            // confidence: seq chunks per batch

typedef __bf16 bf16_t;
typedef bf16_t bf16x8  __attribute__((ext_vector_type(8)));
typedef bf16_t bf16x16 __attribute__((ext_vector_type(16)));
typedef float  f32x8   __attribute__((ext_vector_type(8)));

__device__ __forceinline__ f32x8 wmma_bf16(bf16x16 a, bf16x16 b, f32x8 c) {
  // D = A(16x32 bf16) * B(32x16 bf16) + C(16x16 f32)
  return __builtin_amdgcn_wmma_f32_16x16x32_bf16(
      false, a, false, b, (short)0, c, false, false);
}

// CDNA5 async global->LDS copy (16B per lane), tracked by ASYNCcnt.
__device__ __forceinline__ void async_copy_b128(unsigned lds_off, const void* gaddr) {
  asm volatile("global_load_async_to_lds_b128 %0, %1, off"
               :: "v"(lds_off), "v"((uint64_t)(uintptr_t)gaddr)
               : "memory");
}
__device__ __forceinline__ void wait_async0() {
  asm volatile("s_wait_asynccnt 0x0" ::: "memory");
}

// ---------------- fp32 -> bf16 transpose (weights) ----------------
// in: [K,N] f32 row-major -> out: [N,K] bf16 row-major
__global__ __launch_bounds__(256)
void cvtT_bf16_kernel(const float* __restrict__ in, bf16_t* __restrict__ out,
                      int K, int N) {
  __shared__ float tile[32][33];
  const int k0 = blockIdx.y * 32;
  const int n0 = blockIdx.x * 32;
  const int tx = threadIdx.x & 31;
  const int ty = threadIdx.x >> 5;   // 0..7
#pragma unroll
  for (int i = 0; i < 32; i += 8)
    tile[ty + i][tx] = in[(size_t)(k0 + ty + i) * N + n0 + tx];
  __syncthreads();
#pragma unroll
  for (int i = 0; i < 32; i += 8)
    out[(size_t)(n0 + ty + i) * K + k0 + tx] = (bf16_t)tile[tx][ty + i];
}

// ---------------- LayerNorm (f32 in, bf16 out) ----------------
__global__ __launch_bounds__(256)
void ln_bf16_kernel(const float* __restrict__ x, const float* __restrict__ w,
                    const float* __restrict__ b, bf16_t* __restrict__ out) {
  __shared__ float rs[256];
  __shared__ float rss[256];
  const int row = blockIdx.x;
  const int t = threadIdx.x;
  const float* xr = x + (size_t)row * Dd;
  float vals[4];
  float s = 0.f, ss = 0.f;
#pragma unroll
  for (int i = 0; i < 4; ++i) {
    float v = xr[t + i * 256];
    vals[i] = v;
    s += v;
    ss += v * v;
  }
  rs[t] = s; rss[t] = ss;
  for (int st = 128; st > 0; st >>= 1) {
    __syncthreads();
    if (t < st) { rs[t] += rs[t + st]; rss[t] += rss[t + st]; }
  }
  __syncthreads();
  const float mu = rs[0] * (1.f / Dd);
  const float var = rss[0] * (1.f / Dd) - mu * mu;
  const float inv = rsqrtf(var + 1e-5f);
#pragma unroll
  for (int i = 0; i < 4; ++i) {
    int c = t + i * 256;
    out[(size_t)row * Dd + c] = (bf16_t)((vals[i] - mu) * inv * w[c] + b[c]);
  }
}

// ---------------- bf16 WMMA GEMM with transposed weights ----------------
// C[M,N] = A[M,K] @ B[K,N], B given as BT[N,K] (bf16, row-major).
// block = 256 threads = 8 waves; block tile = 256 rows x 128 cols.
// wave tile = 32 x 128 -> 16 f32x8 accumulators, 16 WMMAs per 32-deep k-step.
// BT tile double-buffered in LDS via CDNA5 async global->LDS (copy/compute overlap).
// OMODE: 0 = bf16 out [M,N]; 1 = bf16 out scattered [B,H,HD,S]; 2 = f32 out [M,N].
template <bool BIAS, bool RESID, bool RELU, int OMODE>
__global__ __launch_bounds__(256)
void gemm_bf16_kernel(const bf16_t* __restrict__ A, const bf16_t* __restrict__ BT,
                      const float* __restrict__ bias, const float* __restrict__ resid,
                      void* __restrict__ outP, int M, int N, int K) {
  __shared__ __align__(16) bf16_t ldsBt[2][128 * 32];  // 2 x 8 KB: [ncol][k]
  const int nc    = blockIdx.x * 128;
  const int mBase = blockIdx.y * 256;
  const int t    = threadIdx.x;
  const int lane = t & 31;
  const int wave = t >> 5;
  const int lh   = lane & 15;
  const int sel  = lane >> 4;
  const int mw   = mBase + wave * 32;

  const f32x8 zero8 = {0.f, 0.f, 0.f, 0.f, 0.f, 0.f, 0.f, 0.f};
  f32x8 acc[2][8];
#pragma unroll
  for (int mh = 0; mh < 2; ++mh)
#pragma unroll
    for (int g = 0; g < 8; ++g) acc[mh][g] = zero8;

  // cooperative staging coords: thread -> (column n, 16-wide k chunk as 2 x b128)
  const int stN = t >> 1;          // 0..127
  const int stK = (t & 1) * 16;    // 0 or 16
  const unsigned ldsBase0 = (unsigned)(uintptr_t)&ldsBt[0][0];
  const unsigned ldsBase1 = (unsigned)(uintptr_t)&ldsBt[1][0];
  const unsigned stOff = (unsigned)((stN * 32 + stK) * 2);

  auto stage = [&](int buf, int kb) {
    const bf16_t* g0 = &BT[(size_t)(nc + stN) * K + kb + stK];
    const unsigned l0 = (buf ? ldsBase1 : ldsBase0) + stOff;
    async_copy_b128(l0, g0);
    async_copy_b128(l0 + 16, g0 + 8);
    if (kb + 32 < K)
      __builtin_prefetch(&BT[(size_t)(nc + stN) * K + kb + 32 + stK], 0, 1);
  };

  stage(0, 0);  // prologue: first tile in flight

  for (int kb = 0; kb < K; kb += 32) {
    const int cur = (kb >> 5) & 1;
    wait_async0();        // our copies for buf[cur] are done
    __syncthreads();      // everyone's copies done + prior reads of buf[cur^1] done
    if (kb + 32 < K) stage(cur ^ 1, kb + 32);   // overlap next copy with compute

    // A fragments: two 16x32 tiles (rows mw..+16, mw+16..+32)
    bf16x16 af[2];
#pragma unroll
    for (int mh = 0; mh < 2; ++mh) {
      const bf16_t* arow = A + (size_t)(mw + mh * 16 + lh) * K + kb + sel * 8;
      bf16x8 lo = *(const bf16x8*)arow;
      bf16x8 hi = *(const bf16x8*)(arow + 16);
#pragma unroll
      for (int e = 0; e < 8; ++e) { af[mh][e] = lo[e]; af[mh][8 + e] = hi[e]; }
    }

    const bf16_t* bbuf = &ldsBt[cur][0];
#pragma unroll
    for (int g = 0; g < 8; ++g) {
      // B fragment: contiguous ds_load_b128 pair from transposed LDS tile
      const bf16_t* brow = bbuf + (g * 16 + lh) * 32 + sel * 8;
      bf16x8 lo = *(const bf16x8*)brow;
      bf16x8 hi = *(const bf16x8*)(brow + 16);
      bf16x16 bfr;
#pragma unroll
      for (int e = 0; e < 8; ++e) { bfr[e] = lo[e]; bfr[8 + e] = hi[e]; }
      acc[0][g] = wmma_bf16(af[0], bfr, acc[0][g]);
      acc[1][g] = wmma_bf16(af[1], bfr, acc[1][g]);
    }
  }

  // epilogue: reg r of acc[mh][g] -> (row = mw + mh*16 + sel*8 + r, col = nc + g*16 + lh)
#pragma unroll
  for (int mh = 0; mh < 2; ++mh) {
#pragma unroll
    for (int g = 0; g < 8; ++g) {
#pragma unroll
      for (int r = 0; r < 8; ++r) {
        const int row = mw + mh * 16 + sel * 8 + r;
        const int col = nc + g * 16 + lh;
        const size_t idx = (size_t)row * N + col;
        float v = acc[mh][g][r];
        if (BIAS)  v += bias[col];
        if (RESID) v += resid[idx];
        if (RELU)  v = v > 0.f ? v : 0.f;
        if (OMODE == 0) {
          ((bf16_t*)outP)[idx] = (bf16_t)v;
        } else if (OMODE == 1) {  // [B,H,HD,S] scatter (V projection)
          const int hh = col >> 6, hd = col & 63;
          const int bb = row >> 11, s = row & (Ssz - 1);
          ((bf16_t*)outP)[(((size_t)bb * Hh + hh) * HDh + hd) * Ssz + s] = (bf16_t)v;
        } else {
          ((float*)outP)[idx] = v;
        }
      }
    }
  }
}

// ---------------- causal flash attention (bf16 WMMA) ----------------
// one wave per (b, h, 16-row query tile); blocks of 4 waves share (b, qt),
// each wave handles a different head -> uniform trip counts for barriers.
// vt is [B,H,HD,S] so P@V B-fragments are contiguous b128 loads.
__global__ __launch_bounds__(128)
void attn_kernel(const bf16_t* __restrict__ q, const bf16_t* __restrict__ k,
                 const bf16_t* __restrict__ vt, bf16_t* __restrict__ ctx) {
  __shared__ __align__(16) bf16_t ldsP[4][16 * 32];  // per-wave P staging (4 KB)
  const int t    = threadIdx.x;
  const int lane = t & 31;
  const int wave = t >> 5;
  const int lh   = lane & 15;
  const int sel  = lane >> 4;

  const int bid = blockIdx.x;          // 0 .. 2047
  const int qt  = bid & 127;           // query tile (S/16)
  const int hg  = (bid >> 7) & 3;      // head group
  const int b   = bid >> 9;            // batch
  const int h   = hg * 4 + wave;
  const size_t rowBase = (size_t)b * Ssz;
  const int hoff = h * HDh;
  const bf16_t* vbase = vt + ((size_t)b * Hh + h) * HDh * Ssz;

  // Q fragments (16x64 split into two 16x32 A-frags), contiguous b128 loads
  bf16x16 qf[2];
  {
    const bf16_t* qrow = q + (rowBase + qt * 16 + lh) * (size_t)Dd + hoff + sel * 8;
#pragma unroll
    for (int j = 0; j < 2; ++j) {
      bf16x8 lo = *(const bf16x8*)(qrow + j * 32);
      bf16x8 hi = *(const bf16x8*)(qrow + j * 32 + 16);
#pragma unroll
      for (int e = 0; e < 8; ++e) { qf[j][e] = lo[e]; qf[j][8 + e] = hi[e]; }
    }
  }

  const f32x8 zero8 = {0.f, 0.f, 0.f, 0.f, 0.f, 0.f, 0.f, 0.f};
  f32x8 ctxa[4] = {zero8, zero8, zero8, zero8};
  float mrow[8], lrow[8];
#pragma unroll
  for (int r = 0; r < 8; ++r) { mrow[r] = -1e30f; lrow[r] = 0.f; }

  const int nchunk = (qt >> 1) + 1;    // 32-key chunks needed for causal rows
  for (int ch = 0; ch < nchunk; ++ch) {
    const int tok0 = ch * 32;

    // ---- scores: two 16x16 subtiles, each = 2 WMMAs over head dim (K=64)
    f32x8 sc[2];
#pragma unroll
    for (int j = 0; j < 2; ++j) {
      const int s0 = tok0 + j * 16;
      const bf16_t* krow = k + (rowBase + s0 + lh) * (size_t)Dd + hoff + sel * 8;
      f32x8 a = zero8;
#pragma unroll
      for (int c = 0; c < 2; ++c) {
        bf16x8 lo = *(const bf16x8*)(krow + c * 32);
        bf16x8 hi = *(const bf16x8*)(krow + c * 32 + 16);
        bf16x16 kf;
#pragma unroll
        for (int e = 0; e < 8; ++e) { kf[e] = lo[e]; kf[8 + e] = hi[e]; }
        a = wmma_bf16(qf[c], kf, a);
      }
      sc[j] = a;
    }

    // ---- scale + causal mask + chunk row-max
    float cmax[8];
#pragma unroll
    for (int r = 0; r < 8; ++r) cmax[r] = -1e30f;
#pragma unroll
    for (int j = 0; j < 2; ++j) {
      const int colTok = tok0 + j * 16 + lh;
#pragma unroll
      for (int r = 0; r < 8; ++r) {
        const int rowTok = qt * 16 + sel * 8 + r;
        float s = sc[j][r] * 0.125f;                 // 1/sqrt(64)
        s = (colTok <= rowTok) ? s : -1e30f;
        sc[j][r] = s;
        cmax[r] = fmaxf(cmax[r], s);
      }
    }
#pragma unroll
    for (int m = 1; m < 16; m <<= 1)
#pragma unroll
      for (int r = 0; r < 8; ++r)
        cmax[r] = fmaxf(cmax[r], __shfl_xor(cmax[r], m));

    float scale[8], csum[8];
#pragma unroll
    for (int r = 0; r < 8; ++r) {
      const float mnew = fmaxf(mrow[r], cmax[r]);
      scale[r] = __expf(mrow[r] - mnew);
      mrow[r] = mnew;
      csum[r] = 0.f;
    }

    // ---- P = exp(s - m), staged through LDS to reshape into A layout
    __syncthreads();
#pragma unroll
    for (int j = 0; j < 2; ++j)
#pragma unroll
      for (int r = 0; r < 8; ++r) {
        const float p = __expf(sc[j][r] - mrow[r]);
        csum[r] += p;
        ldsP[wave][(sel * 8 + r) * 32 + j * 16 + lh] = (bf16_t)p;
      }
#pragma unroll
    for (int m = 1; m < 16; m <<= 1)
#pragma unroll
      for (int r = 0; r < 8; ++r) csum[r] += __shfl_xor(csum[r], m);
#pragma unroll
    for (int r = 0; r < 8; ++r) lrow[r] = lrow[r] * scale[r] + csum[r];
#pragma unroll
    for (int g = 0; g < 4; ++g)
#pragma unroll
      for (int r = 0; r < 8; ++r) ctxa[g][r] *= scale[r];
    __syncthreads();

    // P as one 16x32 A fragment (contiguous b128 LDS reads)
    bf16x16 pf;
    {
      bf16x8 plo = *(const bf16x8*)&ldsP[wave][lh * 32 + sel * 8];
      bf16x8 phi = *(const bf16x8*)&ldsP[wave][lh * 32 + 16 + sel * 8];
#pragma unroll
      for (int e = 0; e < 8; ++e) { pf[e] = plo[e]; pf[8 + e] = phi[e]; }
    }

    // ---- ctx += P(16x32) @ V(32x64): 4 WMMAs, V frags contiguous from vt
#pragma unroll
    for (int g = 0; g < 4; ++g) {
      const bf16_t* vrow = vbase + (size_t)(g * 16 + lh) * Ssz + tok0 + sel * 8;
      bf16x8 lo = *(const bf16x8*)vrow;
      bf16x8 hi = *(const bf16x8*)(vrow + 16);
      bf16x16 vf;
#pragma unroll
      for (int e = 0; e < 8; ++e) { vf[e] = lo[e]; vf[8 + e] = hi[e]; }
      ctxa[g] = wmma_bf16(pf, vf, ctxa[g]);
    }
  }

  // ---- normalize and write ctx (bf16, [B,S,H*HD] head-major)
  float rinv[8];
#pragma unroll
  for (int r = 0; r < 8; ++r) rinv[r] = 1.f / lrow[r];
#pragma unroll
  for (int g = 0; g < 4; ++g) {
#pragma unroll
    for (int r = 0; r < 8; ++r) {
      const int rowTok = qt * 16 + sel * 8 + r;
      ctx[(rowBase + rowTok) * (size_t)Dd + hoff + g * 16 + lh] =
          (bf16_t)(ctxa[g][r] * rinv[r]);
    }
  }
}

// ---------------- confidence head (coalesced two-stage) ----------------
// stage 1: per (b, seq-chunk) partial column sums, fully coalesced reads
__global__ __launch_bounds__(256)
void conf_partial_kernel(const float* __restrict__ xo, float* __restrict__ partial) {
  const int blk = blockIdx.x;            // 0 .. Bsz*CPB-1
  const int t = threadIdx.x;
  const int srows = Ssz / CPB;           // 64
  const float* base = xo + (size_t)blk * srows * Dd;
  float acc[4] = {0.f, 0.f, 0.f, 0.f};
  for (int s = 0; s < srows; ++s) {
    const float* row = base + (size_t)s * Dd;
#pragma unroll
    for (int i = 0; i < 4; ++i) acc[i] += row[t + i * 256];
  }
  float* pp = partial + (size_t)blk * Dd;
#pragma unroll
  for (int i = 0; i < 4; ++i) pp[t + i * 256] = acc[i];
}

// stage 2: deterministic reduce over chunks, dot with conf_w, sigmoid
__global__ __launch_bounds__(256)
void conf_reduce_kernel(const float* __restrict__ partial, const float* __restrict__ cw,
                        const float* __restrict__ cb, float* __restrict__ part) {
  __shared__ float red[256];
  const int b = blockIdx.x;
  const int t = threadIdx.x;
  float acc = 0.f;
  for (int d = t; d < Dd; d += 256) {
    float s = 0.f;
    for (int c = 0; c < CPB; ++c) s += partial[(size_t)(b * CPB + c) * Dd + d];
    acc += (s * (1.f / Ssz)) * cw[d];
  }
  red[t] = acc;
  for (int st = 128; st > 0; st >>= 1) {
    __syncthreads();
    if (t < st) red[t] += red[t + st];
  }
  __syncthreads();
  if (t == 0) part[b] = 1.f / (1.f + __expf(-(red[0] + cb[0])));
}

__global__ void conf_final_kernel(const float* __restrict__ part,
                                  float* __restrict__ dst) {
  dst[0] = 0.25f * (part[0] + part[1] + part[2] + part[3]);
}

// ---------------- host launcher ----------------
extern "C" void kernel_launch(void* const* d_in, const int* in_sizes, int n_in,
                              void* d_out, int out_size, void* d_ws, size_t ws_size,
                              hipStream_t stream) {
  const float* x     = (const float*)d_in[0];
  const float* Wq    = (const float*)d_in[1];
  const float* Wk    = (const float*)d_in[2];
  const float* Wv    = (const float*)d_in[3];
  const float* Wo    = (const float*)d_in[4];
  const float* ln1w  = (const float*)d_in[5];
  const float* ln1b  = (const float*)d_in[6];
  const float* W1    = (const float*)d_in[7];
  const float* b1    = (const float*)d_in[8];
  const float* W2    = (const float*)d_in[9];
  const float* b2    = (const float*)d_in[10];
  const float* ln2w  = (const float*)d_in[11];
  const float* ln2b  = (const float*)d_in[12];
  const float* confw = (const float*)d_in[13];
  const float* confb = (const float*)d_in[14];
  float* out = (float*)d_out;

  uint8_t* ws = (uint8_t*)d_ws;
  size_t off = 0;
  auto alloc = [&](size_t bytes) -> void* {
    void* p = ws + off;
    off += (bytes + 255) & ~(size_t)255;
    return p;
  };
  bf16_t* hbf   = (bf16_t*)alloc((size_t)BS * Dd * 2);
  bf16_t* qbf   = (bf16_t*)alloc((size_t)BS * Dd * 2);
  bf16_t* kbf   = (bf16_t*)alloc((size_t)BS * Dd * 2);
  bf16_t* vtbf  = (bf16_t*)alloc((size_t)BS * Dd * 2);   // [B,H,HD,S]
  bf16_t* ctxbf = (bf16_t*)alloc((size_t)BS * Dd * 2);
  float*  x1    = (float*) alloc((size_t)BS * Dd * 4);
  bf16_t* h2bf  = (bf16_t*)alloc((size_t)BS * Dd * 2);
  bf16_t* ffnbf = (bf16_t*)alloc((size_t)BS * Ff * 2);
  bf16_t* WqT   = (bf16_t*)alloc((size_t)Dd * Dd * 2);   // [N,K] bf16
  bf16_t* WkT   = (bf16_t*)alloc((size_t)Dd * Dd * 2);
  bf16_t* WvT   = (bf16_t*)alloc((size_t)Dd * Dd * 2);
  bf16_t* WoT   = (bf16_t*)alloc((size_t)Dd * Dd * 2);
  bf16_t* W1T   = (bf16_t*)alloc((size_t)Dd * Ff * 2);   // [F,D]
  bf16_t* W2T   = (bf16_t*)alloc((size_t)Ff * Dd * 2);   // [D,F]
  float*  cpartial = (float*)alloc((size_t)Bsz * CPB * Dd * 4);  // 512 KB
  float*  cpart = (float*) alloc(64);

  // 1) weights -> bf16, transposed to [N,K]
  dim3 gT_DD(Dd / 32, Dd / 32);
  cvtT_bf16_kernel<<<gT_DD, 256, 0, stream>>>(Wq, WqT, Dd, Dd);
  cvtT_bf16_kernel<<<gT_DD, 256, 0, stream>>>(Wk, WkT, Dd, Dd);
  cvtT_bf16_kernel<<<gT_DD, 256, 0, stream>>>(Wv, WvT, Dd, Dd);
  cvtT_bf16_kernel<<<gT_DD, 256, 0, stream>>>(Wo, WoT, Dd, Dd);
  dim3 gT_DF(Ff / 32, Dd / 32);            // W1 [D,F] -> [F,D]
  cvtT_bf16_kernel<<<gT_DF, 256, 0, stream>>>(W1, W1T, Dd, Ff);
  dim3 gT_FD(Dd / 32, Ff / 32);            // W2 [F,D] -> [D,F]
  cvtT_bf16_kernel<<<gT_FD, 256, 0, stream>>>(W2, W2T, Ff, Dd);

  // 2) LN1
  ln_bf16_kernel<<<BS, 256, 0, stream>>>(x, ln1w, ln1b, hbf);

  // 3) QKV projections (WMMA GEMM); V stored transposed [B,H,HD,S]
  dim3 gDD(Dd / 128, BS / 256);   // (8, 32)
  gemm_bf16_kernel<false, false, false, 0><<<gDD, 256, 0, stream>>>(
      hbf, WqT, nullptr, nullptr, qbf, BS, Dd, Dd);
  gemm_bf16_kernel<false, false, false, 0><<<gDD, 256, 0, stream>>>(
      hbf, WkT, nullptr, nullptr, kbf, BS, Dd, Dd);
  gemm_bf16_kernel<false, false, false, 1><<<gDD, 256, 0, stream>>>(
      hbf, WvT, nullptr, nullptr, vtbf, BS, Dd, Dd);

  // 4) causal attention (flash-style, WMMA)
  attn_kernel<<<Bsz * (Hh / 4) * (Ssz / 16), 128, 0, stream>>>(qbf, kbf, vtbf, ctxbf);

  // 5) output projection + residual -> x1 (f32)
  gemm_bf16_kernel<false, true, false, 2><<<gDD, 256, 0, stream>>>(
      ctxbf, WoT, nullptr, x, x1, BS, Dd, Dd);

  // 6) LN2 -> h2 (bf16)
  ln_bf16_kernel<<<BS, 256, 0, stream>>>(x1, ln2w, ln2b, h2bf);

  // 7) FFN
  dim3 gDF(Ff / 128, BS / 256);   // (32, 32)
  gemm_bf16_kernel<true, false, true, 0><<<gDF, 256, 0, stream>>>(
      h2bf, W1T, b1, nullptr, ffnbf, BS, Ff, Dd);
  gemm_bf16_kernel<true, true, false, 2><<<gDD, 256, 0, stream>>>(
      ffnbf, W2T, b2, x1, out, BS, Dd, Ff);

  // 8) confidence head
  conf_partial_kernel<<<Bsz * CPB, 256, 0, stream>>>(out, cpartial);
  conf_reduce_kernel<<<Bsz, 256, 0, stream>>>(cpartial, confw, confb, cpart);
  conf_final_kernel<<<1, 1, 0, stream>>>(cpart, out + (size_t)BS * Dd);
}
